// ProjectedMHAMLayer_73194832659005
// MI455X (gfx1250) — matrix operand order repaired and checked by reading, not compiled, not run on hardware
//
#include <hip/hip_runtime.h>

// ---------------------------------------------------------------------------
// ProjectedMHAMLayer for MI455X (gfx1250, wave32, WMMA + async global->LDS)
// ---------------------------------------------------------------------------

#ifndef USE_ASYNC
#define USE_ASYNC 1            // async global->LDS staging (ASYNCcnt path)
#endif

typedef __attribute__((ext_vector_type(16))) __bf16 v16bf;
typedef __attribute__((ext_vector_type(8)))  __bf16 v8bf;
typedef __attribute__((ext_vector_type(8)))  float  v8f;

#define HH 5
#define BB 8
#define LL 20480
#define DD 1024
#define MM 1024
#define NTOK 8192              // BB * MM tokens
#define CONCAT (HH * DD)       // 5120

__device__ __forceinline__ __bf16 f32_to_bf16(float f) {
  unsigned u = __builtin_bit_cast(unsigned, f);
  u += 0x7FFFu + ((u >> 16) & 1u);                 // round-to-nearest-even
  unsigned short s = (unsigned short)(u >> 16);
  return __builtin_bit_cast(__bf16, s);
}

__device__ __forceinline__ float mishf(float x) {
  float sp = (x > 20.f) ? x : log1pf(expf(x));     // softplus
  return x * tanhf(sp);
}

__device__ __forceinline__ float block_sum(float* red, float v, int t) {
  red[t] = v; __syncthreads();
  #pragma unroll
  for (int o = 128; o > 0; o >>= 1) { if (t < o) red[t] += red[t + o]; __syncthreads(); }
  float r = red[0]; __syncthreads();
  return r;
}

__device__ __forceinline__ float block_max(float* red, float v, int t) {
  red[t] = v; __syncthreads();
  #pragma unroll
  for (int o = 128; o > 0; o >>= 1) { if (t < o) red[t] = fmaxf(red[t], red[t + o]); __syncthreads(); }
  float r = red[0]; __syncthreads();
  return r;
}

#if USE_ASYNC
// LDS byte offset of a generic pointer to __shared__ data (addrspace(3) base = 0)
__device__ __forceinline__ unsigned lds_offset(const void* p) {
  return (unsigned)(uintptr_t)(__attribute__((address_space(3))) const void*)p;
}
// GLOBAL_LOAD_ASYNC_TO_LDS_B128: per-lane 16B, tracked by ASYNCcnt (ISA 15.18.3 op 98)
__device__ __forceinline__ void async_b128(unsigned lds_off, const void* gptr) {
  asm volatile("global_load_async_to_lds_b128 %0, %1, off"
               :: "v"(lds_off), "v"((unsigned long long)(uintptr_t)gptr)
               : "memory");
}
__device__ __forceinline__ void wait_async0() {
  asm volatile("s_wait_asynccnt 0" ::: "memory");
}
#endif

// ---------------------------------------------------------------------------
// Fused depthwise-conv (k=3, stride p) + exact avg-pool to M=1024 + input LN,
// for ALL 5 heads in one pass over `value` (p*F == 20 for every head).
// One block per (b,m) row.
// ---------------------------------------------------------------------------
__global__ __launch_bounds__(256) void conv_pool_ln_kernel(
    const float* __restrict__ x, const float* __restrict__ cw,
    const float* __restrict__ cb, const float* __restrict__ g,
    const float* __restrict__ beta, __bf16* __restrict__ projbf,
    float* __restrict__ projlast)
{
  __shared__ float red[256];
  const int row = blockIdx.x;
  const int b = row >> 10, m = row & 1023;
  const int t = threadIdx.x;
  const int P[HH] = {2, 4, 5, 10, 20};
  float pooled[HH][4];

  #pragma unroll
  for (int sub = 0; sub < 4; ++sub) {
    const int d = t + sub * 256;
    const float* xp = x + ((size_t)b * LL + (size_t)m * 20) * DD + d;  // l = m*20
    float xv[21];                                   // l = m*20-1 .. m*20+19
    xv[0] = (m == 0) ? 0.f : xp[-(int)DD];          // left zero-pad at l=-1
    #pragma unroll
    for (int i = 1; i < 21; ++i) xv[i] = xp[(size_t)(i - 1) * DD];
    #pragma unroll
    for (int h = 0; h < HH; ++h) {
      const float w0 = cw[(h * DD + d) * 3 + 0];
      const float w1 = cw[(h * DD + d) * 3 + 1];
      const float w2 = cw[(h * DD + d) * 3 + 2];
      const int p = P[h], F = 20 / p;
      float acc = 0.f;
      for (int j = 0; j < F; ++j)
        acc += w0 * xv[p * j] + w1 * xv[p * j + 1] + w2 * xv[p * j + 2];
      pooled[h][sub] = acc / (float)F + cb[h * DD + d];
    }
  }

  for (int h = 0; h < HH; ++h) {
    float s = 0.f, s2 = 0.f;
    #pragma unroll
    for (int sub = 0; sub < 4; ++sub) { float v = pooled[h][sub]; s += v; s2 += v * v; }
    const float sum   = block_sum(red, s,  t);
    const float sumsq = block_sum(red, s2, t);
    const float mean = sum * (1.f / 1024.f);
    const float inv  = rsqrtf(sumsq * (1.f / 1024.f) - mean * mean + 1e-5f);
    #pragma unroll
    for (int sub = 0; sub < 4; ++sub) {
      const int d = t + sub * 256;
      const float y = (pooled[h][sub] - mean) * inv * g[d] + beta[d];
      projbf[(size_t)h * NTOK * DD + (size_t)row * DD + d] = f32_to_bf16(y);
      if (h == HH - 1) projlast[(size_t)row * DD + d] = y;  // loop-leak residual
    }
  }
}

// ---------------------------------------------------------------------------
// f32 (K x N) row-major  ->  bf16 (N x K) row-major (WMMA B-operand layout)
// ---------------------------------------------------------------------------
__global__ __launch_bounds__(256) void transpose_to_bf16_kernel(
    const float* __restrict__ src, __bf16* __restrict__ dst, int K, int N)
{
  __shared__ float tile[32][33];
  const int n0 = blockIdx.x * 32, k0 = blockIdx.y * 32;
  const int tx = threadIdx.x, ty = threadIdx.y;
  #pragma unroll
  for (int r = 0; r < 4; ++r)
    tile[ty + r * 8][tx] = src[(size_t)(k0 + ty + r * 8) * N + n0 + tx];
  __syncthreads();
  #pragma unroll
  for (int r = 0; r < 4; ++r)
    dst[(size_t)(n0 + ty + r * 8) * K + k0 + tx] = f32_to_bf16(tile[tx][ty + r * 8]);
}

// ---------------------------------------------------------------------------
// bf16 WMMA GEMM:  C(MxN) = epilogue( A(MxK) @ Bt(NxK)^T )
//   workgroup tile 128x128, 8 waves (4 in M x 2 in N), per-wave 32x64
//   (2x4 16x16x32 frags => 8 WMMA / 12 LDS-b128 per K-step), K-step 32,
//   double-buffered LDS (+8 bf16 pad), async global->LDS staging.
// Epilogue modes:
//   0: C f32  (+bias)(+residual)            1: C2 bf16 (+bias)
//   2: C2 bf16 transposed (+bias)           3: C f32 AND C2 = bf16(mish(C))
//   4: C2 = bf16(mish(acc+bias))
// ---------------------------------------------------------------------------
template<int MODE>
__global__ __launch_bounds__(256) void gemm_bf16_kernel(
    const __bf16* __restrict__ A, const __bf16* __restrict__ B,
    float* __restrict__ C, __bf16* __restrict__ C2,
    const float* __restrict__ bias, const float* __restrict__ scale_ptr,
    float scale_imm, const float* __restrict__ residual,
    int K, int lda, int ldb, int ldc, int ldc2,
    long long strideA, long long strideB, long long strideC, long long strideC2)
{
  constexpr int BK = 32, PAD = 8, LDW = BK + PAD;       // 40 bf16 = 80 B rows
  __shared__ __bf16 As[2][128][LDW];
  __shared__ __bf16 Bs[2][128][LDW];

  const int z = blockIdx.z;
  A += (size_t)((long long)z * strideA);
  B += (size_t)((long long)z * strideB);
  const int tileM = blockIdx.y * 128, tileN = blockIdx.x * 128;
  const int t = threadIdx.x;
  const int wid = t >> 5, lane = t & 31, lm = lane & 15, gg = lane >> 4;
  const int wm = wid & 3, wn = wid >> 2;

  v8f acc[2][4] = {};

  // cooperative tile loads: thread t covers rows lr and lr+64, 8-elem chunk lc
  const int lr = t >> 2, lc = (t & 3) * 8;
  const __bf16* Ag0 = A + (size_t)(tileM + lr) * lda + lc;
  const __bf16* Ag1 = Ag0 + (size_t)64 * lda;
  const __bf16* Bg0 = B + (size_t)(tileN + lr) * ldb + lc;
  const __bf16* Bg1 = Bg0 + (size_t)64 * ldb;

#if USE_ASYNC
  const unsigned aoff = lds_offset(&As[0][lr][lc]);
  const unsigned boff = lds_offset(&Bs[0][lr][lc]);
  constexpr unsigned BUFSZ = 128 * LDW * 2;             // 10240 B per buffer
  constexpr unsigned ROW64 = 64 * LDW * 2;              // 5120 B for +64 rows
  auto stage_tile = [&](int buf, int k0) {
    const unsigned ab = aoff + (unsigned)buf * BUFSZ;
    const unsigned bb = boff + (unsigned)buf * BUFSZ;
    async_b128(ab,         Ag0 + k0);
    async_b128(ab + ROW64, Ag1 + k0);
    async_b128(bb,         Bg0 + k0);
    async_b128(bb + ROW64, Bg1 + k0);
  };
  stage_tile(0, 0);
  wait_async0();
  __syncthreads();
#else
  uint4 ra0, ra1, rb0, rb1;
  auto gload = [&](int k0) {
    ra0 = *(const uint4*)(Ag0 + k0);  ra1 = *(const uint4*)(Ag1 + k0);
    rb0 = *(const uint4*)(Bg0 + k0);  rb1 = *(const uint4*)(Bg1 + k0);
  };
  auto sstore = [&](int buf) {
    *(uint4*)&As[buf][lr][lc]      = ra0;
    *(uint4*)&As[buf][lr + 64][lc] = ra1;
    *(uint4*)&Bs[buf][lr][lc]      = rb0;
    *(uint4*)&Bs[buf][lr + 64][lc] = rb1;
  };
  gload(0); sstore(0); __syncthreads();
#endif

  const int nk = K / BK;
  for (int kt = 0; kt < nk; ++kt) {
    const int buf = kt & 1;
#if USE_ASYNC
    if (kt + 1 < nk) stage_tile((kt + 1) & 1, (kt + 1) * BK);
#else
    if (kt + 1 < nk) gload((kt + 1) * BK);
#endif

    v16bf af[2], bfr[4];
    #pragma unroll
    for (int fm = 0; fm < 2; ++fm) {
      // A frag (ISA 7.12.2): lanes 0-15 k={0..7,16..23}, lanes 16-31 k={8..15,24..31}
      const __bf16* p = &As[buf][wm * 32 + fm * 16 + lm][0];
      v8bf lo = *(const v8bf*)(p + 8 * gg);
      v8bf hi = *(const v8bf*)(p + 16 + 8 * gg);
      af[fm] = __builtin_shufflevector(lo, hi, 0,1,2,3,4,5,6,7,8,9,10,11,12,13,14,15);
    }
    #pragma unroll
    for (int fn = 0; fn < 4; ++fn) {
      // B frag: lanes 0-15 k={0..7,16..23}, lanes 16-31 k={8..15,24..31} (n = lane&15)
      const __bf16* p = &Bs[buf][wn * 64 + fn * 16 + lm][8 * gg];
      v8bf lo = *(const v8bf*)(p);
      v8bf hi = *(const v8bf*)(p + 16);
      bfr[fn] = __builtin_shufflevector(lo, hi, 0,1,2,3,4,5,6,7,8,9,10,11,12,13,14,15);
    }
    #pragma unroll
    for (int fm = 0; fm < 2; ++fm)
      #pragma unroll
      for (int fn = 0; fn < 4; ++fn)
        acc[fm][fn] = __builtin_amdgcn_wmma_f32_16x16x32_bf16(
            false, af[fm], false, bfr[fn], (short)0, acc[fm][fn], false, false);

    if (kt + 1 < nk) {
#if USE_ASYNC
      wait_async0();
      __syncthreads();
#else
      sstore((kt + 1) & 1);
      __syncthreads();
#endif
    }
  }

  float sc = scale_imm;
  if (scale_ptr) sc *= scale_ptr[0];
  #pragma unroll
  for (int fm = 0; fm < 2; ++fm) {
    const int row0 = tileM + wm * 32 + fm * 16 + gg * 8;   // C: VGPR r -> M=r+8g
    #pragma unroll
    for (int fn = 0; fn < 4; ++fn) {
      const int col = tileN + wn * 64 + fn * 16 + lm;       // C: N = lane&15
      const float bv = bias ? bias[col] : 0.f;
      #pragma unroll
      for (int r = 0; r < 8; ++r) {
        const int row = row0 + r;
        float v = acc[fm][fn][r] * sc + bv;
        if constexpr (MODE == 0) {
          const size_t ci = (size_t)((long long)z * strideC) + (size_t)row * ldc + col;
          if (residual) v += residual[ci];
          C[ci] = v;
        } else if constexpr (MODE == 1) {
          C2[(size_t)((long long)z * strideC2) + (size_t)row * ldc2 + col] = f32_to_bf16(v);
        } else if constexpr (MODE == 2) {
          C2[(size_t)((long long)z * strideC2) + (size_t)col * ldc2 + row] = f32_to_bf16(v);
        } else if constexpr (MODE == 3) {
          const size_t ci = (size_t)((long long)z * strideC) + (size_t)row * ldc + col;
          C[ci] = v;
          C2[(size_t)((long long)z * strideC2) + (size_t)row * ldc2 + col] = f32_to_bf16(mishf(v));
        } else {
          C2[(size_t)((long long)z * strideC2) + (size_t)row * ldc2 + col] = f32_to_bf16(mishf(v));
        }
      }
    }
  }
}

// ---------------------------------------------------------------------------
// Row softmax over 1024, in place (f32, straight into d_out) + bf16 copy.
// ---------------------------------------------------------------------------
__global__ __launch_bounds__(256) void softmax_kernel(
    float* __restrict__ s, __bf16* __restrict__ ob)
{
  __shared__ float red[256];
  const int row = blockIdx.x, t = threadIdx.x;
  float* p = s + (size_t)row * MM;
  float v[4];
  #pragma unroll
  for (int i = 0; i < 4; ++i) v[i] = p[t + i * 256];
  const float mx = block_max(red, fmaxf(fmaxf(v[0], v[1]), fmaxf(v[2], v[3])), t);
  float sum = 0.f;
  #pragma unroll
  for (int i = 0; i < 4; ++i) { v[i] = expf(v[i] - mx); sum += v[i]; }
  const float inv = 1.f / block_sum(red, sum, t);
  #pragma unroll
  for (int i = 0; i < 4; ++i) {
    const float o = v[i] * inv;
    p[t + i * 256] = o;
    ob[(size_t)row * MM + t + i * 256] = f32_to_bf16(o);
  }
}

// ---------------------------------------------------------------------------
// FF tail: t = v_one + mish(h2raw); LN(ff_ln); LN(norm); -> bf16 concat column.
// ---------------------------------------------------------------------------
__global__ __launch_bounds__(256) void ff_finalize_kernel(
    const float* __restrict__ h2raw, const float* __restrict__ vone,
    const float* __restrict__ ffg, const float* __restrict__ ffb,
    const float* __restrict__ ng, const float* __restrict__ nb,
    __bf16* __restrict__ cat, int hcol)
{
  __shared__ float red[256];
  const int row = blockIdx.x, t = threadIdx.x;
  float v[4];
  float s = 0.f, s2 = 0.f;
  #pragma unroll
  for (int i = 0; i < 4; ++i) {
    const int d = t + i * 256;
    const float x = vone[(size_t)row * DD + d] + mishf(h2raw[(size_t)row * DD + d]);
    v[i] = x; s += x; s2 += x * x;
  }
  float sum = block_sum(red, s, t), sumsq = block_sum(red, s2, t);
  float mean = sum * (1.f / 1024.f);
  float inv  = rsqrtf(sumsq * (1.f / 1024.f) - mean * mean + 1e-5f);
  s = 0.f; s2 = 0.f;
  #pragma unroll
  for (int i = 0; i < 4; ++i) {
    const int d = t + i * 256;
    const float y = (v[i] - mean) * inv * ffg[d] + ffb[d];
    v[i] = y; s += y; s2 += y * y;
  }
  sum = block_sum(red, s, t); sumsq = block_sum(red, s2, t);
  mean = sum * (1.f / 1024.f);
  inv  = rsqrtf(sumsq * (1.f / 1024.f) - mean * mean + 1e-5f);
  #pragma unroll
  for (int i = 0; i < 4; ++i) {
    const int d = t + i * 256;
    cat[(size_t)row * CONCAT + hcol + d] = f32_to_bf16((v[i] - mean) * inv * ng[d] + nb[d]);
  }
}

// ---------------------------------------------------------------------------
extern "C" void kernel_launch(void* const* d_in, const int* in_sizes, int n_in,
                              void* d_out, int out_size, void* d_ws, size_t ws_size,
                              hipStream_t stream)
{
  (void)in_sizes; (void)n_in; (void)out_size; (void)ws_size;
  const float* value   = (const float*)d_in[0];
  const float* conv_w  = (const float*)d_in[1];
  const float* conv_b  = (const float*)d_in[2];
  const float* q_w     = (const float*)d_in[3];
  const float* q_b     = (const float*)d_in[4];
  const float* k_w     = (const float*)d_in[5];
  const float* k_b     = (const float*)d_in[6];
  const float* v_w     = (const float*)d_in[7];
  const float* v_b     = (const float*)d_in[8];
  const float* scale   = (const float*)d_in[9];
  const float* ff_w1   = (const float*)d_in[10];
  const float* ff_b1   = (const float*)d_in[11];
  const float* ff_w2   = (const float*)d_in[12];
  const float* ff_b2   = (const float*)d_in[13];
  const float* ff_ln_g = (const float*)d_in[14];
  const float* ff_ln_b = (const float*)d_in[15];
  const float* norm_g  = (const float*)d_in[16];
  const float* norm_b  = (const float*)d_in[17];
  const float* nin_g   = (const float*)d_in[18];
  const float* nin_b   = (const float*)d_in[19];
  const float* proj_w  = (const float*)d_in[20];
  const float* proj_b  = (const float*)d_in[21];

  float* out      = (float*)d_out;
  float* attn_out = out + (size_t)NTOK * DD;       // (H, B, M, M) f32

  size_t off = 0;
  auto carve = [&](size_t bytes) -> void* {
    void* p = (char*)d_ws + off;
    off += (bytes + 255) & ~(size_t)255;
    return p;
  };
  __bf16* projbf   = (__bf16*)carve((size_t)HH * NTOK * DD * 2);
  float*  projlast = (float*) carve((size_t)NTOK * DD * 4);
  __bf16* qwT      = (__bf16*)carve((size_t)HH * DD * DD * 2);
  __bf16* kwT      = (__bf16*)carve((size_t)HH * DD * DD * 2);
  __bf16* vwT      = (__bf16*)carve((size_t)HH * DD * DD * 2);
  __bf16* w1T      = (__bf16*)carve((size_t)DD * DD * 2);
  __bf16* w2T      = (__bf16*)carve((size_t)DD * DD * 2);
  __bf16* pwT      = (__bf16*)carve((size_t)DD * CONCAT * 2);
  __bf16* Qb       = (__bf16*)carve((size_t)NTOK * DD * 2);
  __bf16* Kb       = (__bf16*)carve((size_t)NTOK * DD * 2);
  __bf16* Vt       = (__bf16*)carve((size_t)NTOK * DD * 2);
  __bf16* attnbf   = (__bf16*)carve((size_t)NTOK * MM * 2);
  float*  vone     = (float*) carve((size_t)NTOK * DD * 4);
  __bf16* mishA    = (__bf16*)carve((size_t)NTOK * DD * 2);
  __bf16* h1m      = (__bf16*)carve((size_t)NTOK * DD * 2);
  float*  h2raw    = (float*) carve((size_t)NTOK * DD * 4);
  __bf16* catb     = (__bf16*)carve((size_t)NTOK * CONCAT * 2);

  // 1) fused conv + pool + input LayerNorm for all heads (one pass over value)
  conv_pool_ln_kernel<<<NTOK, 256, 0, stream>>>(value, conv_w, conv_b, nin_g, nin_b,
                                                projbf, projlast);

  // 2) weights -> bf16 NxK
  const dim3 tb(32, 8);
  for (int h = 0; h < HH; ++h) {
    transpose_to_bf16_kernel<<<dim3(32, 32), tb, 0, stream>>>(q_w + (size_t)h * DD * DD, qwT + (size_t)h * DD * DD, DD, DD);
    transpose_to_bf16_kernel<<<dim3(32, 32), tb, 0, stream>>>(k_w + (size_t)h * DD * DD, kwT + (size_t)h * DD * DD, DD, DD);
    transpose_to_bf16_kernel<<<dim3(32, 32), tb, 0, stream>>>(v_w + (size_t)h * DD * DD, vwT + (size_t)h * DD * DD, DD, DD);
  }
  transpose_to_bf16_kernel<<<dim3(32, 32),  tb, 0, stream>>>(ff_w1,  w1T, DD, DD);
  transpose_to_bf16_kernel<<<dim3(32, 32),  tb, 0, stream>>>(ff_w2,  w2T, DD, DD);
  transpose_to_bf16_kernel<<<dim3(32, 160), tb, 0, stream>>>(proj_w, pwT, CONCAT, DD);

  const float inv_sqrt = 0.03125f;              // 1/sqrt(1024)
  const dim3 gFull(DD / 128, NTOK / 128, 1);    // (8, 64, 1)
  const dim3 gBatch(DD / 128, MM / 128, BB);    // (8, 8, 8)
  const long long TD = (long long)MM * DD;      // per-batch token*feature stride

  for (int h = 0; h < HH; ++h) {
    const __bf16* Ph = projbf + (size_t)h * NTOK * DD;
    // Q, K  (bf16 out; K's natural token-major layout IS the B-operand layout)
    gemm_bf16_kernel<1><<<gFull, 256, 0, stream>>>(Ph, qwT + (size_t)h * DD * DD, nullptr, Qb,
        q_b + h * DD, nullptr, 1.f, nullptr, DD, DD, DD, DD, DD, 0, 0, 0, 0);
    gemm_bf16_kernel<1><<<gFull, 256, 0, stream>>>(Ph, kwT + (size_t)h * DD * DD, nullptr, Kb,
        k_b + h * DD, nullptr, 1.f, nullptr, DD, DD, DD, DD, DD, 0, 0, 0, 0);
    // V, written transposed per batch -> Vt[b][dq][m]
    gemm_bf16_kernel<2><<<gBatch, 256, 0, stream>>>(Ph, vwT + (size_t)h * DD * DD, nullptr, Vt,
        v_b + h * DD, nullptr, 1.f, nullptr, DD, DD, DD, DD, MM, TD, 0, 0, TD);
    // scores = Q K^T / 32, f32 straight into d_out attn region
    gemm_bf16_kernel<0><<<gBatch, 256, 0, stream>>>(Qb, Kb,
        attn_out + (size_t)h * BB * MM * MM, nullptr,
        nullptr, nullptr, inv_sqrt, nullptr, DD, DD, DD, MM, 0, TD, TD, (long long)MM * MM, 0);
    // softmax in place + bf16 copy
    softmax_kernel<<<NTOK, 256, 0, stream>>>(attn_out + (size_t)h * BB * MM * MM, attnbf);
    // v_one = scale[h] * attn @ V   (f32 residual copy + bf16(mish) for FF)
    gemm_bf16_kernel<3><<<gBatch, 256, 0, stream>>>(attnbf, Vt, vone, mishA,
        nullptr, scale + h, 1.f, nullptr, MM, MM, MM, DD, DD,
        (long long)MM * MM, TD, TD, TD);
    // h1 = mish(mishA @ W1 + b1)
    gemm_bf16_kernel<4><<<gFull, 256, 0, stream>>>(mishA, w1T, nullptr, h1m,
        ff_b1, nullptr, 1.f, nullptr, DD, DD, DD, DD, DD, 0, 0, 0, 0);
    // h2raw = h1 @ W2 + b2
    gemm_bf16_kernel<0><<<gFull, 256, 0, stream>>>(h1m, w2T, h2raw, nullptr,
        ff_b2, nullptr, 1.f, nullptr, DD, DD, DD, DD, 0, 0, 0, 0, 0);
    // residual + mish + LN(ff) + LN(norm) -> concat column block h (bf16)
    ff_finalize_kernel<<<NTOK, 256, 0, stream>>>(h2raw, vone, ff_ln_g, ff_ln_b,
                                                 norm_g, norm_b, catb, h * DD);
  }

  // out = concat @ proj_w + proj_b + projected(last head)
  gemm_bf16_kernel<0><<<gFull, 256, 0, stream>>>(catb, pwT, out, nullptr,
      proj_b, nullptr, 1.f, projlast, CONCAT, CONCAT, CONCAT, DD, 0, 0, 0, 0, 0);
}
